// deformable_LKA_experimental_1881195676324
// MI455X (gfx1250) — matrix operand order repaired and checked by reading, not compiled, not run on hardware
//
#include <hip/hip_runtime.h>
#include <hip/hip_bf16.h>
#include <stdint.h>
#include <stddef.h>

// ---------------- constants for this problem ----------------
constexpr int BN = 4, CD = 256, HH = 64, WWD = 64;
constexpr int HWX = HH * WWD;        // 4096
constexpr int MROWS = BN * HWX;      // 16384 pixel-rows
constexpr int KDIM = 256;            // GEMM K (channels)

typedef __attribute__((ext_vector_type(16))) _Float16 v16h;
typedef __attribute__((ext_vector_type(8)))  _Float16 v8h;
typedef __attribute__((ext_vector_type(8)))  float    v8f;

// ---------------- 1) NCHW -> NHWC transpose (fp32 + f16) ----------------
__global__ __launch_bounds__(256) void transpose_x_kernel(
    const float* __restrict__ x, float* __restrict__ xf32,
    _Float16* __restrict__ xf16) {
  __shared__ float tl[32][33];
  int t   = threadIdx.x;
  int bid = blockIdx.x;          // 4 * 128 * 8 = 4096 blocks
  int b   = bid >> 10;           // 1024 tiles per batch
  int tb  = bid & 1023;
  int hw0 = (tb & 127) * 32;     // 128 hw-tiles
  int c0  = (tb >> 7) * 32;      // 8 c-tiles
  const float* xb = x + (size_t)b * CD * HWX;
  int cw  = t >> 5;              // 0..7
  int hwl = t & 31;
#pragma unroll
  for (int i = 0; i < 4; ++i) {
    int cl = cw + i * 8;
    tl[cl][hwl] = xb[(size_t)(c0 + cl) * HWX + hw0 + hwl];
  }
  __syncthreads();
  int cl2 = t & 31;
  int hwr = t >> 5;
#pragma unroll
  for (int i = 0; i < 4; ++i) {
    int hwl2 = hwr + i * 8;
    float v  = tl[cl2][hwl2];
    size_t o = ((size_t)b * HWX + hw0 + hwl2) * CD + c0 + cl2;
    xf32[o] = v;
    xf16[o] = (_Float16)v;
  }
}

// ---------------- 2) weight repack [O,I] fp32 -> [K, Npad] f16 ----------------
__global__ __launch_bounds__(256) void convert_w_kernel(
    const float* __restrict__ w, _Float16* __restrict__ dst,
    int Nout, int Npad) {
  int idx = blockIdx.x * 256 + threadIdx.x;       // grid sized exactly K*Npad/256
  int k = idx / Npad, n = idx % Npad;
  float v = (n < Nout) ? w[(size_t)n * KDIM + k] : 0.0f;
  dst[idx] = (_Float16)v;
}

// ---------------- 3) WMMA GEMM: [M,256]f16 x [256,Npad]f16 + bias -> fp32 ----
__global__ __launch_bounds__(256) void gemm16_kernel(
    const _Float16* __restrict__ A, const _Float16* __restrict__ Bw,
    const float* __restrict__ bias, float* __restrict__ Cout,
    int Npad, int Nout) {
  int lane = threadIdx.x & 31;
  int wave = threadIdx.x >> 5;
  int h = lane >> 4, l16 = lane & 15;
  int m0 = blockIdx.x * 128 + wave * 16;
  int n0 = blockIdx.y * 16;
  int n  = n0 + l16;
  float bv = (n < Nout) ? bias[n] : 0.0f;
  v8f c = {bv, bv, bv, bv, bv, bv, bv, bv};
  const _Float16* ap = A + (size_t)(m0 + l16) * KDIM + 8 * h;      // 16B aligned
  const _Float16* bp = Bw + (size_t)(8 * h) * Npad + n0 + l16;     // column n0+l16, K offset 8h
#pragma unroll
  for (int k0 = 0; k0 < KDIM; k0 += 32) {
    v8h alo = *(const v8h*)(ap + k0);         // K = k0+8h .. +7
    v8h ahi = *(const v8h*)(ap + k0 + 16);    // K = k0+16+8h .. +7
    v16h a = __builtin_shufflevector(alo, ahi, 0, 1, 2, 3, 4, 5, 6, 7,
                                     8, 9, 10, 11, 12, 13, 14, 15);
    v16h bfrag;  // K = k0+8h+s (s<8), k0+16+8h+(s-8) (s>=8)
#pragma unroll
    for (int s = 0; s < 8; ++s) {
      bfrag[s] = bp[(size_t)(k0 + s) * Npad];
      bfrag[s + 8] = bp[(size_t)(k0 + 16 + s) * Npad];
    }
    c = __builtin_amdgcn_wmma_f32_16x16x32_f16(false, a, false, bfrag,
                                               (short)0, c, false, false);
  }
  if (n < Nout) {
#pragma unroll
    for (int r = 0; r < 8; ++r)
      Cout[(size_t)(m0 + r + 8 * h) * Nout + n] = c[r];
  }
}

// ---------------- 4) depthwise 3x3, pad 1, channels-last ----------------
__global__ __launch_bounds__(256) void dwconv3_kernel(
    const float* __restrict__ in, const float* __restrict__ w,
    const float* __restrict__ bias, float* __restrict__ out, int Cc, int total) {
  int idx = blockIdx.x * 256 + threadIdx.x;
  if (idx >= total) return;
  int c = idx % Cc;
  int p = idx / Cc;
  int b = p >> 12, yx = p & 4095;
  int y = yx >> 6, x = yx & 63;
  float acc = bias[c];
  const float* wc = w + c * 9;
#pragma unroll
  for (int i = 0; i < 3; ++i) {
    int yy = y + i - 1;
    if (yy < 0 || yy >= HH) continue;
#pragma unroll
    for (int j = 0; j < 3; ++j) {
      int xx = x + j - 1;
      if (xx < 0 || xx >= WWD) continue;
      acc += in[((size_t)((b << 12) + (yy << 6) + xx)) * Cc + c] * wc[i * 3 + j];
    }
  }
  out[idx] = acc;
}

// ---------------- 5) deformable depthwise conv, shared offsets ----------------
template <int K, int PAD, int DIL>
__global__ __launch_bounds__(256) void deform_kernel(
    const float* __restrict__ src,     // NHWC fp32
    const float* __restrict__ off,     // [p][2*K*K]
    const float* __restrict__ dw,      // [C][K*K]
    float* __restrict__ dst32,         // NHWC fp32 (optional)
    _Float16* __restrict__ dst16) {    // NHWC f16
  constexpr int K2 = K * K;
  __shared__ float s_w[K2][4];
  __shared__ int   s_i[K2][4];
  int p = blockIdx.x;
  int b = p >> 12, yx = p & 4095;
  int y = yx >> 6, x = yx & 63;
  int c = threadIdx.x;
  if (c < K2) {
    int k = c;
    float dy = off[(size_t)p * (2 * K2) + 2 * k];
    float dx = off[(size_t)p * (2 * K2) + 2 * k + 1];
    float py = (float)y + (float)((k / K) * DIL - PAD) + dy;
    float px = (float)x + (float)((k % K) * DIL - PAD) + dx;
    float y0f = floorf(py), x0f = floorf(px);
    float ty = py - y0f, tx = px - x0f;
    int y0 = (int)y0f, x0 = (int)x0f;
#pragma unroll
    for (int q = 0; q < 4; ++q) {
      int yi = y0 + (q >> 1), xi = x0 + (q & 1);
      float wq = ((q >> 1) ? ty : 1.0f - ty) * ((q & 1) ? tx : 1.0f - tx);
      bool ok = (yi >= 0) && (yi < HH) && (xi >= 0) && (xi < WWD);
      s_w[k][q] = wq;
      s_i[k][q] = ok ? (yi * WWD + xi) : -1;
    }
  }
  __syncthreads();
  float acc = 0.0f;
  const float* srcb = src + (((size_t)b) << 12) * CD + c;
  const float* dwc  = dw + c * K2;
#pragma unroll 1
  for (int k = 0; k < K2; ++k) {
    float s = 0.0f;
#pragma unroll
    for (int q = 0; q < 4; ++q) {
      int idx = s_i[k][q];                       // uniform across the block
      if (idx >= 0) s += srcb[(size_t)idx * CD] * s_w[k][q];
    }
    acc += s * dwc[k];
  }
  size_t o = (size_t)p * CD + c;
  if (dst32) dst32[o] = acc;
  dst16[o] = (_Float16)acc;
}

// ---------------- 6) final WMMA GEMM fused with x*attn, NCHW store ----------
__global__ __launch_bounds__(256) void gemm_final_kernel(
    const _Float16* __restrict__ A, const _Float16* __restrict__ Bw,
    const float* __restrict__ bias, const float* __restrict__ xin,
    float* __restrict__ out) {
  __shared__ float tile[8][16 * 17];
  constexpr int NP = 256;
  int lane = threadIdx.x & 31;
  int wave = threadIdx.x >> 5;
  int h = lane >> 4, l16 = lane & 15;
  int m0 = blockIdx.x * 128 + wave * 16;
  int n0 = blockIdx.y * 16;
  float bv = bias[n0 + l16];
  v8f c = {bv, bv, bv, bv, bv, bv, bv, bv};
  const _Float16* ap = A + (size_t)(m0 + l16) * KDIM + 8 * h;
  const _Float16* bp = Bw + (size_t)(8 * h) * NP + n0 + l16;
#pragma unroll
  for (int k0 = 0; k0 < KDIM; k0 += 32) {
    v8h alo = *(const v8h*)(ap + k0);
    v8h ahi = *(const v8h*)(ap + k0 + 16);
    v16h a = __builtin_shufflevector(alo, ahi, 0, 1, 2, 3, 4, 5, 6, 7,
                                     8, 9, 10, 11, 12, 13, 14, 15);
    v16h bfrag;
#pragma unroll
    for (int s = 0; s < 8; ++s) {
      bfrag[s] = bp[(size_t)(k0 + s) * NP];
      bfrag[s + 8] = bp[(size_t)(k0 + 16 + s) * NP];
    }
    c = __builtin_amdgcn_wmma_f32_16x16x32_f16(false, a, false, bfrag,
                                               (short)0, c, false, false);
  }
  float* t = tile[wave];
#pragma unroll
  for (int r = 0; r < 8; ++r) t[(r + 8 * h) * 17 + l16] = c[r];
  __syncthreads();
  int b = m0 >> 12;
  int yx0 = m0 & 4095;
#pragma unroll
  for (int i = 0; i < 8; ++i) {
    int e = i * 32 + lane;
    int nl = e >> 4, ml = e & 15;
    size_t oidx = ((size_t)(b * CD + n0 + nl)) * HWX + (size_t)(yx0 + ml);
    out[oidx] = t[ml * 17 + nl] * xin[oidx];   // 64B-contiguous per half-wave
  }
}

// ---------------- host launcher ----------------
extern "C" void kernel_launch(void* const* d_in, const int* in_sizes, int n_in,
                              void* d_out, int out_size, void* d_ws, size_t ws_size,
                              hipStream_t stream) {
  (void)in_sizes; (void)n_in; (void)out_size; (void)ws_size;
  const float* x      = (const float*)d_in[0];
  const float* adj_w0 = (const float*)d_in[1];
  const float* adj_b0 = (const float*)d_in[2];
  const float* off_w0 = (const float*)d_in[3];
  const float* off_b0 = (const float*)d_in[4];
  const float* dw0    = (const float*)d_in[5];
  const float* adj_w1 = (const float*)d_in[6];
  const float* adj_b1 = (const float*)d_in[7];
  const float* off_w1 = (const float*)d_in[8];
  const float* off_b1 = (const float*)d_in[9];
  const float* dw1    = (const float*)d_in[10];
  const float* w1     = (const float*)d_in[11];
  const float* b1     = (const float*)d_in[12];
  float* out = (float*)d_out;

  // workspace layout (bytes, all offsets 256B-aligned)
  char* ws = (char*)d_ws;
  float*    xf32  = (float*)(ws + 0);                       // 16,777,216
  _Float16* af16  = (_Float16*)(ws + 16777216);             //  8,388,608 (x/attn f16)
  float*    attnA = (float*)(ws + 25165824);                // 16,777,216
  float*    xc    = (float*)(ws + 41943040);                //  7,340,032
  float*    offb  = (float*)(ws + 49283072);                //  7,340,032
  _Float16* w0h   = (_Float16*)(ws + 56623104);             //     32,768 (256x64)
  _Float16* w1h   = (_Float16*)(ws + 56655872);             //     57,344 (256x112)
  _Float16* wfh   = (_Float16*)(ws + 56713216);             //    131,072 (256x256)

  // 1) transpose x -> NHWC fp32 + f16
  transpose_x_kernel<<<4096, 256, 0, stream>>>(x, xf32, af16);
  // 2) weight repacks
  convert_w_kernel<<<(KDIM * 64) / 256, 256, 0, stream>>>(adj_w0, w0h, 50, 64);
  convert_w_kernel<<<(KDIM * 112) / 256, 256, 0, stream>>>(adj_w1, w1h, 98, 112);
  convert_w_kernel<<<(KDIM * 256) / 256, 256, 0, stream>>>(w1, wfh, 256, 256);

  // ---- block 0 (K=5, pad=2, dil=1) ----
  gemm16_kernel<<<dim3(MROWS / 128, 4), 256, 0, stream>>>(af16, w0h, adj_b0, xc, 64, 50);
  {
    int total = MROWS * 50;
    dwconv3_kernel<<<(total + 255) / 256, 256, 0, stream>>>(xc, off_w0, off_b0, offb, 50, total);
  }
  deform_kernel<5, 2, 1><<<MROWS, 256, 0, stream>>>(xf32, offb, dw0, attnA, af16);

  // ---- block 1 (K=7, pad=9, dil=3) ----
  gemm16_kernel<<<dim3(MROWS / 128, 7), 256, 0, stream>>>(af16, w1h, adj_b1, xc, 112, 98);
  {
    int total = MROWS * 98;
    dwconv3_kernel<<<(total + 255) / 256, 256, 0, stream>>>(xc, off_w1, off_b1, offb, 98, total);
  }
  deform_kernel<7, 9, 3><<<MROWS, 256, 0, stream>>>(attnA, offb, dw1, nullptr, af16);

  // ---- final 1x1 conv + x*attn, fused NCHW epilogue ----
  gemm_final_kernel<<<dim3(MROWS / 128, 16), 256, 0, stream>>>(af16, wfh, b1, x, out);
}
